// Model_60352880443586
// MI455X (gfx1250) — compile-verified, matrix operand order checked by reading, not run on hardware
//
#include <hip/hip_runtime.h>

// ---------------------------------------------------------------------------
// Bidirectional LSTM, T=512, B=64, I=H=512, PyTorch gate order i,f,g,o.
// Batch-split persistent WGs (2 per direction, 32 batch rows each), fused
// [x_t | h] x [W_ih | W_hh]^T GEMM per step via v_wmma_f32_16x16x32_f16.
// 2 M-tiles per WG -> every weight fragment feeds 2 WMMAs (halves L2 ingest).
// f32 cell state in LDS (lane-exclusive), h resident in LDS, no inter-WG sync.
// ---------------------------------------------------------------------------

typedef _Float16 f16x8  __attribute__((ext_vector_type(8)));
typedef _Float16 f16x16 __attribute__((ext_vector_type(16)));
typedef float    f32x8  __attribute__((ext_vector_type(8)));

#define TT 512
#define BB 64
#define II 512
#define HH 512
#define GG 2048          // 4*H gate columns
#define KC 1024          // I + H fused K
#define AP 1032          // padded LDS A row stride (f16 elems): +16B pad
#define CP 516           // padded LDS C row stride (f32 elems)
#define A_BYTES (32 * AP * 2)            // 66048
#define C_BYTES (32 * CP * 4)            // 66048
#define LDS_BYTES (A_BYTES + C_BYTES)    // 132096

__device__ __forceinline__ float sigf(float v) { return 1.0f / (1.0f + __expf(-v)); }

// Convert 32 contiguous f32 -> f16 into dst (dst 16B-aligned).
__device__ __forceinline__ void cvt_store32(_Float16* dst, const float* src) {
  #pragma unroll
  for (int q = 0; q < 4; ++q) {
    float4 a = ((const float4*)src)[2 * q];
    float4 b = ((const float4*)src)[2 * q + 1];
    f16x8 p;
    p[0] = (_Float16)a.x; p[1] = (_Float16)a.y;
    p[2] = (_Float16)a.z; p[3] = (_Float16)a.w;
    p[4] = (_Float16)b.x; p[5] = (_Float16)b.y;
    p[6] = (_Float16)b.z; p[7] = (_Float16)b.w;
    *(f16x8*)(dst + 8 * q) = p;
  }
}

// ---------------------------------------------------------------------------
// Pack W = [W_ih | W_hh] (f32, row-major [2048 x 1024]) into f16 WMMA
// B-fragment lane order so GEMM inner-loop B loads are coalesced b128s.
// Linear order: dir(2) x g_tile(128) x k_blk(32) x half(2) x lane(32) x 8 f16.
// ---------------------------------------------------------------------------
__global__ __launch_bounds__(256)
void pack_w_kernel(const float* __restrict__ wih_f, const float* __restrict__ whh_f,
                   const float* __restrict__ wih_b, const float* __restrict__ whh_b,
                   _Float16* __restrict__ wpack)
{
  unsigned linear = blockIdx.x * blockDim.x + threadIdx.x;   // 0 .. 524287
  unsigned lane = linear & 31u;
  unsigned half = (linear >> 5) & 1u;
  unsigned kb   = (linear >> 6) & 31u;
  unsigned gt   = (linear >> 11) & 127u;
  unsigned dir  = linear >> 18;

  unsigned g  = gt * 16u + (lane & 15u);
  unsigned k0 = kb * 32u + half * 16u + (lane >> 4) * 8u;

  const float* wih = dir ? wih_b : wih_f;
  const float* whh = dir ? whh_b : whh_f;
  const float* src = (k0 < II) ? (wih + (size_t)g * II + k0)
                               : (whh + (size_t)g * HH + (k0 - II));
  const float4 v0 = ((const float4*)src)[0];
  const float4 v1 = ((const float4*)src)[1];
  f16x8 p;
  p[0] = (_Float16)v0.x; p[1] = (_Float16)v0.y;
  p[2] = (_Float16)v0.z; p[3] = (_Float16)v0.w;
  p[4] = (_Float16)v1.x; p[5] = (_Float16)v1.y;
  p[6] = (_Float16)v1.z; p[7] = (_Float16)v1.w;
  *(f16x8*)(wpack + (size_t)linear * 8) = p;
}

// ---------------------------------------------------------------------------
// Persistent LSTM kernel. Grid = 4 blocks (dir = bx>>1, batch tile = bx&1),
// block = 512 threads = 16 waves. WG owns 32 batch rows (2 M-tiles).
// Wave w owns h-columns [w*32, w*32+32): per strip s, g_tile = s*32 + w*2 + u.
// ---------------------------------------------------------------------------
__global__ __launch_bounds__(512)
void lstm_kernel(const float* __restrict__ x,     // [T,B,I]
                 const float* __restrict__ h0,    // [2,B,H]
                 const float* __restrict__ c0,    // [2,B,H]
                 const float* __restrict__ b_f,   // [4H]
                 const float* __restrict__ b_b,   // [4H]
                 const _Float16* __restrict__ wpack,
                 float* __restrict__ out)         // [T,B,2H]
{
  extern __shared__ char smem[];
  _Float16* A = (_Float16*)smem;                  // [32 rows x (x|h) 1024], padded
  float*    C = (float*)(smem + A_BYTES);         // [32 rows x 512 cell], padded

  const int tid   = threadIdx.x;
  const int dir   = blockIdx.x >> 1;
  const int btile = blockIdx.x & 1;
  const int wave  = tid >> 5;                     // 0..15
  const int lane  = tid & 31;
  const int l15   = lane & 15;
  const int lhi   = lane >> 4;                    // 0 or 1
  const int brow0 = btile * 32;                   // global batch row base

  const float*  bias = dir ? b_b : b_f;
  const f16x8*  wp8  = (const f16x8*)(wpack + (size_t)dir * ((size_t)GG * KC));

  const int j0 = wave * 32;                       // h-column window: 32 cols

  // Staging geometry: each thread owns 32 contiguous elements of one row.
  const int srow = tid >> 4;                      // 0..31
  const int scol = (tid & 15) * 32;               // 0..480

  // Per-lane gate biases (constant over the sequence). u = sub-window.
  float bia[2][4];
  #pragma unroll
  for (int u = 0; u < 2; ++u)
    #pragma unroll
    for (int s = 0; s < 4; ++s)
      bia[u][s] = bias[s * HH + j0 + u * 16 + l15];

  // Preload h0 -> LDS A[:,512:1024], c0 -> LDS C.
  cvt_store32(&A[srow * AP + II + scol],
              h0 + ((size_t)dir * BB + brow0 + srow) * HH + scol);
  {
    const float* cs = c0 + ((size_t)dir * BB + brow0 + srow) * HH + scol;
    #pragma unroll
    for (int q = 0; q < 8; ++q)
      *(float4*)&C[srow * CP + scol + 4 * q] = ((const float4*)cs)[q];
  }

  const int arow0 = l15 * AP;                     // A-frag row base, M-tile 0
  const int arow1 = (16 + l15) * AP;              // A-frag row base, M-tile 1

  for (int t = 0; t < TT; ++t) {
    const int tin = dir ? (TT - 1 - t) : t;       // backward runs time-reversed
    const float* xs = x + ((size_t)tin * BB + brow0) * II;

    // Opaque offset: blocks LICM from hoisting t-invariant weight loads out
    // of the t loop (and spilling them), without breaking addrspace inference.
    size_t wofs = 0;
    asm volatile("" : "+s"(wofs));
    const f16x8* wpt = wp8 + wofs;

    // Stage x_t (f32 -> f16) into A[:, 0:512]; h sits in A[:, 512:1024].
    cvt_store32(&A[srow * AP + scol], xs + (size_t)srow * II + scol);
    __syncthreads();

    f32x8 acc[2][2][4];                           // [m][u][s]
    #pragma unroll
    for (int m = 0; m < 2; ++m)
      #pragma unroll
      for (int u = 0; u < 2; ++u)
        #pragma unroll
        for (int s = 0; s < 4; ++s)
          acc[m][u][s] = (f32x8){0.f, 0.f, 0.f, 0.f, 0.f, 0.f, 0.f, 0.f};

    // K loop: per kb issue 4 ds_load_b128 (A, 2 M-tiles) + 16 global b128
    // (8 B fragments), then 16 WMMAs (2 per fragment, consumed in order).
    #pragma unroll 2
    for (int kb = 0; kb < 32; ++kb) {
      const int ka = kb * 32 + lhi * 8;
      f16x8 a0lo = *(const f16x8*)&A[arow0 + ka];
      f16x8 a0hi = *(const f16x8*)&A[arow0 + ka + 16];
      f16x8 a1lo = *(const f16x8*)&A[arow1 + ka];
      f16x8 a1hi = *(const f16x8*)&A[arow1 + ka + 16];
      f16x16 af0 = __builtin_shufflevector(a0lo, a0hi,
                   0, 1, 2, 3, 4, 5, 6, 7, 8, 9, 10, 11, 12, 13, 14, 15);
      f16x16 af1 = __builtin_shufflevector(a1lo, a1hi,
                   0, 1, 2, 3, 4, 5, 6, 7, 8, 9, 10, 11, 12, 13, 14, 15);
      f16x16 bf[8];
      #pragma unroll
      for (int u = 0; u < 2; ++u)
        #pragma unroll
        for (int s = 0; s < 4; ++s) {
          const int gt = s * 32 + wave * 2 + u;
          const size_t blk = ((size_t)gt * 32 + kb) * 64;   // in f16x8 units
          f16x8 blo = wpt[blk + lane];
          f16x8 bhi = wpt[blk + 32 + lane];
          bf[u * 4 + s] = __builtin_shufflevector(blo, bhi,
                          0, 1, 2, 3, 4, 5, 6, 7, 8, 9, 10, 11, 12, 13, 14, 15);
        }
      #pragma unroll
      for (int u = 0; u < 2; ++u)
        #pragma unroll
        for (int s = 0; s < 4; ++s) {
          acc[0][u][s] = __builtin_amdgcn_wmma_f32_16x16x32_f16(
              false, af0, false, bf[u * 4 + s], (short)0, acc[0][u][s], false, false);
          acc[1][u][s] = __builtin_amdgcn_wmma_f32_16x16x32_f16(
              false, af1, false, bf[u * 4 + s], (short)0, acc[1][u][s], false, false);
        }
    }
    __syncthreads();   // all waves done reading A before h write-back

    // Gate nonlinearities + state update. Each lane exclusively owns its
    // (row,col) cell elements -> no sync needed around C. h -> LDS, f32 -> out.
    #pragma unroll
    for (int m = 0; m < 2; ++m)
      #pragma unroll
      for (int u = 0; u < 2; ++u) {
        const int rb  = m * 16 + lhi * 8;          // local row base for this lane
        const int col = j0 + u * 16 + l15;         // h column 0..511
        #pragma unroll
        for (int e = 0; e < 8; ++e) {
          float ig = sigf(acc[m][u][0][e] + bia[u][0]);
          float fg = sigf(acc[m][u][1][e] + bia[u][1]);
          float gg = tanhf(acc[m][u][2][e] + bia[u][2]);
          float og = sigf(acc[m][u][3][e] + bia[u][3]);
          float cn = fg * C[(rb + e) * CP + col] + ig * gg;
          float hn = og * tanhf(cn);
          C[(rb + e) * CP + col] = cn;
          A[(rb + e) * AP + II + col] = (_Float16)hn;
          out[((size_t)tin * BB + brow0 + rb + e) * (2 * HH) + (size_t)dir * HH + col] = hn;
        }
      }
    // Next iteration's post-stage __syncthreads orders h writes vs. GEMM reads;
    // x staging (cols < 512) is disjoint from h writes (cols >= 512).
  }
}

extern "C" void kernel_launch(void* const* d_in, const int* in_sizes, int n_in,
                              void* d_out, int out_size, void* d_ws, size_t ws_size,
                              hipStream_t stream) {
  const float* x     = (const float*)d_in[0];
  const float* h0    = (const float*)d_in[1];
  const float* c0    = (const float*)d_in[2];
  const float* wih_f = (const float*)d_in[3];
  const float* whh_f = (const float*)d_in[4];
  const float* b_f   = (const float*)d_in[5];
  const float* wih_b = (const float*)d_in[6];
  const float* whh_b = (const float*)d_in[7];
  const float* b_b   = (const float*)d_in[8];
  float* out = (float*)d_out;

  _Float16* wpack = (_Float16*)d_ws;   // 2 * 2048 * 1024 * 2B = 8 MB

  // Phase 1: pack weights into WMMA B-fragment layout (524288 threads).
  pack_w_kernel<<<2048, 256, 0, stream>>>(wih_f, whh_f, wih_b, whh_b, wpack);

  // Phase 2: persistent recurrent kernel. 2 dirs x 2 batch tiles, 16 waves,
  // 129 KB dynamic LDS (A + cell state) per WG.
  lstm_kernel<<<4, 512, LDS_BYTES, stream>>>(x, h0, c0, b_f, b_b, wpack, out);
}
